// Attention_63823214019267
// MI455X (gfx1250) — matrix operand order repaired and checked by reading, not compiled, not run on hardware
//
#include <hip/hip_runtime.h>
#include <hip/hip_bf16.h>
#include <math.h>

// ---------------------------------------------------------------------------
// MI455X (gfx1250): all heavy math through v_wmma_f32_16x16x32_f16; big GEMMs
// (QKV / MLP1 / MLP2 / proj) run block-cooperatively with the B panel staged
// into LDS via global_load_async_to_lds_b128 (ASYNCcnt double-buffering), so
// WMMA consumes ds_load_b128 while the next tile is in flight.
// ---------------------------------------------------------------------------

typedef __attribute__((ext_vector_type(16))) _Float16 v16h;
typedef __attribute__((ext_vector_type(8)))  _Float16 v8h;
typedef __attribute__((ext_vector_type(8)))  float    v8f;

#define DEV __device__ __forceinline__

static constexpr int BB = 32, NN = 320, CC = 768, HH = 12, SS = 256, TT = 64, HD = 64;
static constexpr int BH = BB * HH;          // 384
static constexpr int ROWS = BB * NN;        // 10240
static constexpr int MLPM = BB * SS;        // 8192

DEV v16h make_v16(v8h lo, v8h hi) {
    v16h r;
#pragma unroll
    for (int i = 0; i < 8; ++i) { r[i] = lo[i]; r[i + 8] = hi[i]; }
    return r;
}

// A operand (16x32 f16): lane = M row (lane&15); lanes 0-15 hold K[0..15],
// lanes 16-31 hold K[16..31]  (two 16B contiguous loads per lane)
DEV v16h load_A_tile(const _Float16* __restrict__ A, int lda, int m0, int k0, int lane) {
    const _Float16* p = A + (size_t)(m0 + (lane & 15)) * lda + k0 + ((lane >> 4) * 16);
    v8h lo = *(const v8h*)p;
    v8h hi = *(const v8h*)(p + 8);
    return make_v16(lo, hi);
}

// B operand (32x16 f16) from K-contiguous (transposed) storage Bt[n,k]:
// lane = N column (lane&15); lanes 0-15 hold K[0..15], lanes 16-31 hold K[16..31]
DEV v16h load_B_tile(const _Float16* __restrict__ Bt, int ldb, int n0, int k0, int lane) {
    const _Float16* p = Bt + (size_t)(n0 + (lane & 15)) * ldb + k0 + ((lane >> 4) * 16);
    return *(const v16h*)p;
}

DEV v8f wmma16(v16h a, v16h b, v8f c) {
    return __builtin_amdgcn_wmma_f32_16x16x32_f16(false, a, false, b, (short)0, c,
                                                  false, false);
}

// Async copy of 16B global -> LDS (ASYNCcnt tracked).
DEV void async_g2l_b128(unsigned lds_off, const void* gptr) {
    asm volatile("global_load_async_to_lds_b128 %0, %1, off"
                 :: "v"(lds_off), "v"((unsigned long long)(uintptr_t)gptr)
                 : "memory");
}
DEV void wait_async0() { asm volatile("s_wait_asynccnt 0" ::: "memory"); }
DEV void wait_async1() { asm volatile("s_wait_asynccnt 1" ::: "memory"); }

// ---------------------------------------------------------------------------
__global__ __launch_bounds__(256) void f32_to_f16_kernel(const float* __restrict__ in,
                                                         _Float16* __restrict__ out, int n) {
    int i = blockIdx.x * blockDim.x + threadIdx.x;
    if (i < n) out[i] = (_Float16)in[i];
}

// tgt_rep[b,c] = mean_t(x[b,t,c]*mask[b,t]), t<64 -> f16
__global__ __launch_bounds__(256) void tgt_kernel(const float* __restrict__ x,
                                                  const float* __restrict__ mask,
                                                  _Float16* __restrict__ tgt16) {
    int b = blockIdx.x;
    __shared__ float sm[TT];
    if (threadIdx.x < TT) sm[threadIdx.x] = mask[b * TT + threadIdx.x];
    __syncthreads();
    for (int c = threadIdx.x; c < CC; c += blockDim.x) {
        float acc = 0.f;
#pragma unroll 8
        for (int t = 0; t < TT; ++t)
            acc += x[((size_t)b * NN + t) * CC + c] * sm[t];
        tgt16[b * CC + c] = (_Float16)(acc * (1.0f / (float)TT));
    }
}

// mlp_in[b*256+s, 0:768]=x16[b,64+s,:]; [768:1536]=tgt16[b,:]
__global__ __launch_bounds__(256) void pack_mlp_in_kernel(const _Float16* __restrict__ x16,
                                                          const _Float16* __restrict__ tgt16,
                                                          _Float16* __restrict__ mi) {
    size_t i = (size_t)blockIdx.x * blockDim.x + threadIdx.x;
    if (i >= (size_t)MLPM * 2 * CC) return;
    int col = (int)(i % (2 * CC));
    int row = (int)(i / (2 * CC));
    int b = row / SS, s = row % SS;
    _Float16 v = (col < CC) ? x16[((size_t)b * NN + TT + s) * CC + col]
                            : tgt16[b * CC + (col - CC)];
    mi[i] = v;
}

// ---------------------------------------------------------------------------
// Block-cooperative NT WMMA GEMM. One block (8 waves) computes a 128x64 C tile;
// wave w owns rows m0 = bm*128 + w*16. The shared 64x32 B tile per k-step is
// staged global->LDS with async b128 loads, double-buffered on ASYNCcnt.
// MODE 0: f32 store (+bias). MODE 1: exact GELU, f16 store. MODE 2: QKV scatter.
// ---------------------------------------------------------------------------
template <int MODE>
__global__ __launch_bounds__(256) void gemm_async_kernel(const _Float16* __restrict__ A,
                                                         const _Float16* __restrict__ Bt,
                                                         const float* __restrict__ bias,
                                                         void* __restrict__ Cv,
                                                         _Float16* __restrict__ q,
                                                         _Float16* __restrict__ k,
                                                         _Float16* __restrict__ vT,
                                                         int M, int N, int K) {
    __shared__ __align__(64) _Float16 Bl[2][64][32];
    const int mb = M >> 7;                       // blocks along M (M % 128 == 0)
    const int bm = blockIdx.x % mb;
    const int bn = blockIdx.x / mb;
    const int n0 = bn * 64;
    const int tid = threadIdx.x;
    const int lane = tid & 31;
    const int wv = tid >> 5;
    const int m0 = bm * 128 + wv * 16;
    const int srow = tid >> 2;                   // 0..63: B row staged by this thread
    const int sseg = (tid & 3) * 8;              // 8-half (16B) segment
    const _Float16* bsrc = Bt + (size_t)(n0 + srow) * K + sseg;

    // prologue: stage k-step 0 into buffer 0
    async_g2l_b128((unsigned)(uintptr_t)&Bl[0][srow][sseg], bsrc);

    v8f acc[4] = {};
    const int nk = K >> 5;
    for (int ik = 0; ik < nk; ++ik) {
        const int buf = ik & 1;
        if (ik + 1 < nk) {
            async_g2l_b128((unsigned)(uintptr_t)&Bl[buf ^ 1][srow][sseg],
                           bsrc + (ik + 1) * 32);
            wait_async1();                       // own stage for `buf` complete
        } else {
            wait_async0();
        }
        __syncthreads();                         // all waves' LDS writes visible

        const int k0 = ik * 32;
        if (k0 + 32 < K)
            __builtin_prefetch(A + (size_t)(m0 + (lane & 15)) * K + k0 + 32, 0, 0);
        v16h a = load_A_tile(A, K, m0, k0, lane);
#pragma unroll
        for (int t = 0; t < 4; ++t) {
            const _Float16* bp = &Bl[buf][t * 16 + (lane & 15)][(lane >> 4) * 16];
            v16h b = *(const v16h*)bp;           // 2x ds_load_b128
            acc[t] = wmma16(a, b, acc[t]);
        }
        __syncthreads();                         // buffer safe to overwrite
    }

    const int half = lane >> 4, l16 = lane & 15;
#pragma unroll
    for (int t = 0; t < 4; ++t) {
        const int n = n0 + t * 16 + l16;
        const float bv = bias[n];
        if (MODE == 2) {
            const int which = n / CC;
            const int cc = n % CC;
            const int head = cc >> 6, d = cc & 63;
#pragma unroll
            for (int r = 0; r < 8; ++r) {
                const int m = m0 + r + 8 * half;
                const int b = m / NN, tok = m % NN;
                const int bh = b * HH + head;
                _Float16 hv = (_Float16)(acc[t][r] + bv);
                if (which == 0)      q[((size_t)bh * NN + tok) * HD + d] = hv;
                else if (which == 1) k[((size_t)bh * NN + tok) * HD + d] = hv;
                else                 vT[((size_t)bh * HD + d) * NN + tok] = hv;
            }
        } else {
#pragma unroll
            for (int r = 0; r < 8; ++r) {
                const int m = m0 + r + 8 * half;
                float v = acc[t][r] + bv;
                if (MODE == 1) {
                    v = 0.5f * v * (1.0f + erff(v * 0.70710678118f));
                    ((_Float16*)Cv)[(size_t)m * N + n] = (_Float16)v;
                } else {
                    ((float*)Cv)[(size_t)m * N + n] = v;
                }
            }
        }
    }
}

// MLP layer 3 (K=192,N=2) + argmax -> classes + one-hot decision
__global__ __launch_bounds__(256) void mlp3_kernel(const _Float16* __restrict__ h2,
                                                   const float* __restrict__ w3,
                                                   const float* __restrict__ b3,
                                                   int* __restrict__ classes,
                                                   float* __restrict__ dec_out) {
    int i = blockIdx.x * blockDim.x + threadIdx.x;
    if (i >= ROWS) return;
    int tok = i % NN;
    if (tok < TT) { classes[i] = 0; return; }
    int b = i / NN;
    int row = b * SS + (tok - TT);
    float l0 = b3[0], l1 = b3[1];
    const _Float16* hp = h2 + (size_t)row * 192;
#pragma unroll 4
    for (int k = 0; k < 192; ++k) {
        float hv = (float)hp[k];
        l0 += hv * w3[k];
        l1 += hv * w3[192 + k];
    }
    int am = (l1 > l0) ? 1 : 0;   // jnp.argmax: ties -> first
    classes[i] = 1 + am;
    dec_out[(size_t)row * 2 + 0] = (am == 0) ? 1.f : 0.f;
    dec_out[(size_t)row * 2 + 1] = (am == 1) ? 1.f : 0.f;
}

__global__ __launch_bounds__(64) void vsum_kernel(const _Float16* __restrict__ vT,
                                                  float* __restrict__ vsum) {
    int bh = blockIdx.x, d = threadIdx.x;
    const _Float16* p = vT + ((size_t)bh * HD + d) * NN;
    float acc = 0.f;
#pragma unroll 8
    for (int j = 0; j < NN; ++j) acc += (float)p[j];
    vsum[bh * HD + d] = acc;
}

// Flash attention with policy mask. One wave = one 16-row i-tile of one (b,h).
// Mask: ap(i,j)=0 iff class_i+class_j==1 ; running max over RAW scores (matches
// the reference full-row max); eps terms folded as rank-1 vsum correction.
__global__ __launch_bounds__(256) void attn_kernel(const _Float16* __restrict__ q,
                                                   const _Float16* __restrict__ k,
                                                   const _Float16* __restrict__ vT,
                                                   const int* __restrict__ classes,
                                                   const float* __restrict__ vsum,
                                                   _Float16* __restrict__ hidden) {
    __shared__ __align__(64) _Float16 Plds[8][16][32];
    const float SCALE = 0.125f;           // hd^-0.5
    int lane = threadIdx.x & 31;
    int wiw = threadIdx.x >> 5;
    int wid = blockIdx.x * 8 + wiw;       // 7680 waves exactly
    int itile = wid % 20;
    int bh = wid / 20;
    int b = bh / HH, h = bh % HH;
    int i0 = itile * 16;
    int half = lane >> 4, l16 = lane & 15;

    const _Float16* qb = q + (size_t)bh * NN * HD;
    const _Float16* kb = k + (size_t)bh * NN * HD;
    const _Float16* vb = vT + (size_t)bh * HD * NN;

    v16h aq0 = load_A_tile(qb, HD, i0, 0, lane);
    v16h aq1 = load_A_tile(qb, HD, i0, 32, lane);

    const int* cbase = classes + b * NN;
    int cls_i[8];
#pragma unroll
    for (int r = 0; r < 8; ++r) cls_i[r] = cbase[i0 + r + 8 * half];

    float mx[8], ls[8];
#pragma unroll
    for (int r = 0; r < 8; ++r) { mx[r] = -3.0e38f; ls[r] = 0.f; }
    v8f o[4] = {};

    for (int jt = 0; jt < 10; ++jt) {
        int j0 = jt * 32;
        v8f z = {};
        v8f s0 = wmma16(aq1, load_B_tile(kb, HD, j0, 32, lane),
                        wmma16(aq0, load_B_tile(kb, HD, j0, 0, lane), z));
        v8f s1 = wmma16(aq1, load_B_tile(kb, HD, j0 + 16, 32, lane),
                        wmma16(aq0, load_B_tile(kb, HD, j0 + 16, 0, lane), z));
        int cj0 = cbase[j0 + l16];
        int cj1 = cbase[j0 + 16 + l16];
#pragma unroll
        for (int r = 0; r < 8; ++r) {
            float raw0 = s0[r] * SCALE;
            float raw1 = s1[r] * SCALE;
            float rm = fmaxf(raw0, raw1);
            rm = fmaxf(rm, __shfl_xor(rm, 1, 32));
            rm = fmaxf(rm, __shfl_xor(rm, 2, 32));
            rm = fmaxf(rm, __shfl_xor(rm, 4, 32));
            rm = fmaxf(rm, __shfl_xor(rm, 8, 32));
            float mnew = fmaxf(mx[r], rm);
            float fsc = __expf(mx[r] - mnew);
            int ci = cls_i[r];
            float p0 = ((ci + cj0) != 1) ? __expf(raw0 - mnew) : 0.f;
            float p1 = ((ci + cj1) != 1) ? __expf(raw1 - mnew) : 0.f;
            float rs = p0 + p1;
            rs += __shfl_xor(rs, 1, 32);
            rs += __shfl_xor(rs, 2, 32);
            rs += __shfl_xor(rs, 4, 32);
            rs += __shfl_xor(rs, 8, 32);
            ls[r] = ls[r] * fsc + rs;
            mx[r] = mnew;
#pragma unroll
            for (int dt = 0; dt < 4; ++dt) o[dt][r] *= fsc;
            Plds[wiw][r + 8 * half][l16] = (_Float16)p0;
            Plds[wiw][r + 8 * half][16 + l16] = (_Float16)p1;
        }
        __syncthreads();
        // P-tile (16x32) back as A operand
        const _Float16* pr = &Plds[wiw][l16][0];
        v8h plo = *(const v8h*)(pr + half * 16);
        v8h phi = *(const v8h*)(pr + half * 16 + 8);
        v16h ap = make_v16(plo, phi);
#pragma unroll
        for (int dt = 0; dt < 4; ++dt) {
            v16h bv = load_B_tile(vb, NN, dt * 16, j0, lane);
            o[dt] = wmma16(ap, bv, o[dt]);
        }
        __syncthreads();
    }

    const float eps = 1e-6f, epsn = 1e-6f / (float)NN;
#pragma unroll
    for (int dt = 0; dt < 4; ++dt) {
        float vs = vsum[bh * HD + dt * 16 + l16];
#pragma unroll
        for (int r = 0; r < 8; ++r) {
            float val = (o[dt][r] + epsn * vs) / (ls[r] + eps);
            int tok = i0 + r + 8 * half;
            hidden[((size_t)b * NN + tok) * CC + h * HD + dt * 16 + l16] = (_Float16)val;
        }
    }
}

// ---------------------------------------------------------------------------
extern "C" void kernel_launch(void* const* d_in, const int* in_sizes, int n_in,
                              void* d_out, int out_size, void* d_ws, size_t ws_size,
                              hipStream_t stream) {
    (void)in_sizes; (void)n_in; (void)out_size; (void)ws_size;
    const float* x      = (const float*)d_in[0];
    const float* tmask  = (const float*)d_in[1];
    const float* qkv_w  = (const float*)d_in[2];
    const float* qkv_b  = (const float*)d_in[3];
    const float* proj_w = (const float*)d_in[4];
    const float* proj_b = (const float*)d_in[5];
    const float* dp1_w  = (const float*)d_in[6];
    const float* dp1_b  = (const float*)d_in[7];
    const float* dp2_w  = (const float*)d_in[8];
    const float* dp2_b  = (const float*)d_in[9];
    const float* dp3_w  = (const float*)d_in[10];
    const float* dp3_b  = (const float*)d_in[11];

    char* w = (char*)d_ws;
    size_t off = 0;
    auto alloc = [&](size_t bytes) -> char* {
        char* p = w + off;
        off = (off + bytes + 255) & ~(size_t)255;
        return p;
    };
    _Float16* x16     = (_Float16*)alloc((size_t)ROWS * CC * 2);
    _Float16* qkvw16  = (_Float16*)alloc((size_t)3 * CC * CC * 2);
    _Float16* projw16 = (_Float16*)alloc((size_t)CC * CC * 2);
    _Float16* dp1w16  = (_Float16*)alloc((size_t)384 * 2 * CC * 2);
    _Float16* dp2w16  = (_Float16*)alloc((size_t)192 * 384 * 2);
    _Float16* tgt16   = (_Float16*)alloc((size_t)BB * CC * 2);
    _Float16* mlp_in  = (_Float16*)alloc((size_t)MLPM * 2 * CC * 2);
    _Float16* h1      = (_Float16*)alloc((size_t)MLPM * 384 * 2);
    _Float16* h2      = (_Float16*)alloc((size_t)MLPM * 192 * 2);
    int*      classes = (int*)alloc((size_t)ROWS * 4);
    _Float16* q16     = (_Float16*)alloc((size_t)BH * NN * HD * 2);
    _Float16* k16     = (_Float16*)alloc((size_t)BH * NN * HD * 2);
    _Float16* vT16    = (_Float16*)alloc((size_t)BH * HD * NN * 2);
    float*    vsum    = (float*)alloc((size_t)BH * HD * 4);
    _Float16* hidden  = (_Float16*)alloc((size_t)ROWS * CC * 2);

    float* out_main = (float*)d_out;
    float* dec_out  = (float*)d_out + (size_t)ROWS * CC;

    auto cvt = [&](const float* src, _Float16* dst, int n) {
        f32_to_f16_kernel<<<(n + 255) / 256, 256, 0, stream>>>(src, dst, n);
    };
    cvt(x, x16, ROWS * CC);
    cvt(qkv_w, qkvw16, 3 * CC * CC);
    cvt(proj_w, projw16, CC * CC);
    cvt(dp1_w, dp1w16, 384 * 2 * CC);
    cvt(dp2_w, dp2w16, 192 * 384);

    tgt_kernel<<<BB, 256, 0, stream>>>(x, tmask, tgt16);

    {
        size_t tot = (size_t)MLPM * 2 * CC;
        pack_mlp_in_kernel<<<(int)((tot + 255) / 256), 256, 0, stream>>>(x16, tgt16, mlp_in);
    }

    auto gblocks = [](long M, long N) { return (int)((M / 128) * (N / 64)); };

    gemm_async_kernel<1><<<gblocks(MLPM, 384), 256, 0, stream>>>(
        mlp_in, dp1w16, dp1_b, h1, nullptr, nullptr, nullptr, MLPM, 384, 2 * CC);
    gemm_async_kernel<1><<<gblocks(MLPM, 192), 256, 0, stream>>>(
        h1, dp2w16, dp2_b, h2, nullptr, nullptr, nullptr, MLPM, 192, 384);
    mlp3_kernel<<<(ROWS + 255) / 256, 256, 0, stream>>>(h2, dp3_w, dp3_b, classes, dec_out);

    gemm_async_kernel<2><<<gblocks(ROWS, 3 * CC), 256, 0, stream>>>(
        x16, qkvw16, qkv_b, nullptr, q16, k16, vT16, ROWS, 3 * CC, CC);
    vsum_kernel<<<BH, 64, 0, stream>>>(vT16, vsum);

    attn_kernel<<<(BH * (NN / 16)) / 8, 256, 0, stream>>>(q16, k16, vT16, classes, vsum,
                                                          hidden);

    gemm_async_kernel<0><<<gblocks(ROWS, CC), 256, 0, stream>>>(
        hidden, projw16, proj_b, out_main, nullptr, nullptr, nullptr, ROWS, CC, CC);
}
